// GAT_34900904247864
// MI455X (gfx1250) — matrix operand order repaired and checked by reading, not compiled, not run on hardware
//
#include <hip/hip_runtime.h>
#include <hip/hip_bf16.h>

typedef _Float16 half_t;
typedef __attribute__((ext_vector_type(16))) _Float16 v16h;
typedef __attribute__((ext_vector_type(8)))  float    v8f;

#define NEG_SLOPE 0.2f

// ---------------- float <-> ordered-uint key for atomic max on floats ----------
__device__ __forceinline__ unsigned fkey(float f) {
    unsigned u = __float_as_uint(f);
    return (u & 0x80000000u) ? ~u : (u | 0x80000000u);
}
__device__ __forceinline__ float funkey(unsigned k) {
    return __uint_as_float((k & 0x80000000u) ? (k ^ 0x80000000u) : ~k);
}
#define NEG_INF_KEY 0x007FFFFFu   // fkey(-inf)

// ---------------- utility -------------------------------------------------------
__global__ void fill_u32(unsigned* __restrict__ p, unsigned v, int n) {
    int i = blockIdx.x * blockDim.x + threadIdx.x;
    if (i < n) p[i] = v;
}

// ============== WMMA fragment pre-packing ======================================
// CDNA5 ISA 7.12.2 per-lane layouts for V_WMMA_F32_16X16X32_F16 (wave32):
//   A (16x32): lane l holds row m=l&15;  half i -> K = i + 8*(l>>4) + (i>=8 ? 8:0)
//   B (32x16): lane l holds col n=l&15;  half i -> K = i + 8*(l>>4) + (i>=8 ? 8:0)
//   C/D (16x16 f32): vgpr r -> row r + 8*(l>>4), col l&15
// Packed linear index: ((tile0*nTile1 + tile1)*32 + lane)*16 + i
// so each lane fetches one contiguous 32B v16h per fragment.

__global__ void pack_a_f32(const float* __restrict__ src, half_t* __restrict__ dst,
                           int M, int K) {
    int p = blockIdx.x * blockDim.x + threadIdx.x;
    if (p >= M * K) return;
    int i    = p & 15;
    int lane = (p >> 4) & 31;
    int nKT  = K >> 5;
    int kt   = (p >> 9) % nKT;
    int rt   = (p >> 9) / nKT;
    int hi   = lane >> 4;
    int m    = rt * 16 + (lane & 15);
    int k32  = (i < 8) ? (i + 8 * hi) : (i + 8 + 8 * hi);
    dst[p] = (half_t)src[(size_t)m * K + kt * 32 + k32];
}

__global__ void pack_b_f32(const float* __restrict__ W, half_t* __restrict__ dst,
                           int K, int Nc) {
    int p = blockIdx.x * blockDim.x + threadIdx.x;
    if (p >= K * Nc) return;
    int i    = p & 15;
    int lane = (p >> 4) & 31;
    int nCT  = Nc >> 4;
    int ct   = (p >> 9) % nCT;
    int kt   = (p >> 9) / nCT;
    int hi   = lane >> 4;
    int col  = ct * 16 + (lane & 15);
    int k32  = (i < 8) ? (i + 8 * hi) : (i + 8 + 8 * hi);
    dst[p] = (half_t)W[(size_t)(kt * 32 + k32) * Nc + col];
}

// layer-1 epilogue: +b1, ELU, convert to f16 AND write in packed-A order
__global__ void elu_bias_pack(const float* __restrict__ agg, const float* __restrict__ b,
                              half_t* __restrict__ dst, int M, int K) {
    int p = blockIdx.x * blockDim.x + threadIdx.x;
    if (p >= M * K) return;
    int i    = p & 15;
    int lane = (p >> 4) & 31;
    int nKT  = K >> 5;
    int kt   = (p >> 9) % nKT;
    int rt   = (p >> 9) / nKT;
    int hi   = lane >> 4;
    int m    = rt * 16 + (lane & 15);
    int k32  = (i < 8) ? (i + 8 * hi) : (i + 8 + 8 * hi);
    int c    = kt * 32 + k32;
    float v  = agg[(size_t)m * K + c] + b[c];
    v = v > 0.f ? v : (__expf(v) - 1.f);
    dst[p] = (half_t)v;
}

// ============== packed-fragment WMMA GEMM ======================================
// One wave computes a 16 x (16*CT) tile; A fragment reused across CT B tiles.
// K is compile-time: NKT k-steps of 32, fully unrolled -> NKT*CT back-to-back WMMAs.
// Requires M % 16 == 0 (true here: 50000 = 16*3125) -> unconditional stores.
template<int NKT, int CT>
__global__ void wmma_gemm_packed(const half_t* __restrict__ A, const half_t* __restrict__ B,
                                 float* __restrict__ C, int Nc) {
    const int lane = threadIdx.x & 31;
    const int rt   = blockIdx.x;
    const int nCT  = Nc >> 4;
    const int ct0  = (threadIdx.x >> 5) * CT;
    const int hi   = lane >> 4;
    const int n    = lane & 15;

    const v16h* Ap = (const v16h*)A;
    const v16h* Bp = (const v16h*)B;

    v8f acc[CT];
#pragma unroll
    for (int j = 0; j < CT; ++j) acc[j] = (v8f){};

#pragma unroll
    for (int kt = 0; kt < NKT; ++kt) {
        v16h a = Ap[(size_t)(rt * NKT + kt) * 32 + lane];
#pragma unroll
        for (int j = 0; j < CT; ++j) {
            v16h b = Bp[(size_t)(kt * nCT + ct0 + j) * 32 + lane];
            acc[j] = __builtin_amdgcn_wmma_f32_16x16x32_f16(
                /*neg_a=*/false, a, /*neg_b=*/false, b,
                /*c_mod=*/(short)0, acc[j], /*reuse_a=*/false, /*reuse_b=*/false);
        }
    }
#pragma unroll
    for (int j = 0; j < CT; ++j) {
        int col = (ct0 + j) * 16 + n;
#pragma unroll
        for (int r = 0; r < 8; ++r)
            C[(size_t)(rt * 16 + r + 8 * hi) * Nc + col] = acc[j][r];
    }
}

// ---------------- per-node attention coefficients ------------------------------
__global__ void node_alpha(const float* __restrict__ h, const float* __restrict__ avs,
                           const float* __restrict__ avd, float* __restrict__ asrc,
                           float* __restrict__ adst, int Nn, int heads, int C) {
    int idx = blockIdx.x * blockDim.x + threadIdx.x;
    if (idx >= Nn * heads) return;
    int nd = idx / heads, hd = idx - nd * heads;
    const float* hp = h + (size_t)nd * heads * C + hd * C;
    const float* s  = avs + hd * C;
    const float* d  = avd + hd * C;
    float vs = 0.f, vd = 0.f;
    for (int c = 0; c < C; ++c) { vs += hp[c] * s[c]; vd += hp[c] * d[c]; }
    asrc[idx] = vs;
    adst[idx] = vd;
}

// ---------------- edge pass 1: segment max of leaky-relu logits ----------------
__global__ void edge_max(const int* __restrict__ ei, const float* __restrict__ asrc,
                         const float* __restrict__ adst, unsigned* __restrict__ m,
                         int E, int Nn, int heads) {
    int idx = blockIdx.x * blockDim.x + threadIdx.x;
    int tot = (E + Nn) * heads;
    if (idx >= tot) return;
    int e = idx / heads, hd = idx - e * heads;
    int s, d;
    if (e < E) { s = ei[e]; d = ei[E + e]; } else { s = d = e - E; }
    float lg = asrc[s * heads + hd] + adst[d * heads + hd];
    lg = lg > 0.f ? lg : NEG_SLOPE * lg;
    atomicMax(&m[d * heads + hd], fkey(lg));
}

// ---------------- edge pass 2: segment sum of exp(logit - max) -----------------
__global__ void edge_expsum(const int* __restrict__ ei, const float* __restrict__ asrc,
                            const float* __restrict__ adst, const unsigned* __restrict__ m,
                            float* __restrict__ ssum, int E, int Nn, int heads) {
    int idx = blockIdx.x * blockDim.x + threadIdx.x;
    int tot = (E + Nn) * heads;
    if (idx >= tot) return;
    int e = idx / heads, hd = idx - e * heads;
    int s, d;
    if (e < E) { s = ei[e]; d = ei[E + e]; } else { s = d = e - E; }
    float lg = asrc[s * heads + hd] + adst[d * heads + hd];
    lg = lg > 0.f ? lg : NEG_SLOPE * lg;
    float ev = __expf(lg - funkey(m[d * heads + hd]));
    atomicAdd(&ssum[d * heads + hd], ev);
}

// ---------------- edge pass 3: wave-per-edge weighted scatter-add --------------
template<int HEADS, int CPH>
__global__ void edge_agg(const int* __restrict__ ei, const float* __restrict__ asrc,
                         const float* __restrict__ adst, const unsigned* __restrict__ m,
                         const float* __restrict__ ssum, const float* __restrict__ feat,
                         float* __restrict__ out, int E, int Nn) {
    int gid  = blockIdx.x * blockDim.x + threadIdx.x;
    int lane = threadIdx.x & 31;
    int e    = gid >> 5;                 // uniform within wave
    if (e >= E + Nn) return;
    int s, d;
    if (e < E) { s = ei[e]; d = ei[E + e]; } else { s = d = e - E; }
    const float* fs = feat + (size_t)s * (HEADS * CPH);
    __builtin_prefetch(fs + lane, 0, 0);    // global_prefetch of gathered row
    float alpha = 0.f;
    if (lane < HEADS) {
        float lg = asrc[s * HEADS + lane] + adst[d * HEADS + lane];
        lg = lg > 0.f ? lg : NEG_SLOPE * lg;
        float ev = __expf(lg - funkey(m[d * HEADS + lane]));
        alpha = ev / ssum[d * HEADS + lane];
    }
    float* od = out + (size_t)d * (HEADS * CPH);
#pragma unroll
    for (int j = 0; j < (HEADS * CPH) / 32; ++j) {
        int c  = j * 32 + lane;
        int hd = c / CPH;
        float a = __shfl(alpha, hd, 32);
        atomicAdd(&od[c], a * fs[c]);
    }
}

// ---------------- final: out += b2 ---------------------------------------------
__global__ void bias_add(float* __restrict__ out, const float* __restrict__ b,
                         int total, int ld) {
    int i = blockIdx.x * blockDim.x + threadIdx.x;
    if (i >= total) return;
    out[i] += b[i % ld];
}

// ==============================================================================
extern "C" void kernel_launch(void* const* d_in, const int* in_sizes, int n_in,
                              void* d_out, int out_size, void* d_ws, size_t ws_size,
                              hipStream_t stream) {
    const int IN_DIM = 256, HEADS = 8, HID = 32, OUT_DIM = 64;
    const int HID1 = HEADS * HID;                 // 256
    const int N = in_sizes[0] / IN_DIM;           // 50000  (divisible by 16)
    const int E = in_sizes[1] / 2;                // 800000
    const int ET = E + N;                         // with self-loops

    const float* x     = (const float*)d_in[0];
    const int*   ei    = (const int*)  d_in[1];
    const float* W1    = (const float*)d_in[2];
    const float* asv1  = (const float*)d_in[3];
    const float* adv1  = (const float*)d_in[4];
    const float* b1    = (const float*)d_in[5];
    const float* W2    = (const float*)d_in[6];
    const float* asv2  = (const float*)d_in[7];
    const float* adv2  = (const float*)d_in[8];
    const float* b2    = (const float*)d_in[9];
    float*       out   = (float*)d_out;

    // ---- workspace carve-out (256B aligned) ----
    size_t off = 0;
    char* base = (char*)d_ws;
    auto alloc = [&](size_t bytes) -> void* {
        off = (off + 255) & ~(size_t)255;
        void* p = base + off;
        off += bytes;
        return p;
    };
    float*    h1    = (float*)   alloc((size_t)N * HID1 * 4);   // layer1 GEMM out
    float*    agg1  = (float*)   alloc((size_t)N * HID1 * 4);   // layer1 aggregate
    float*    g2    = (float*)   alloc((size_t)N * OUT_DIM * 4);// layer2 GEMM out
    half_t*   pA1   = (half_t*)  alloc((size_t)N * IN_DIM * 2); // packed A (x)
    half_t*   pB1   = (half_t*)  alloc((size_t)IN_DIM * HID1 * 2);
    half_t*   pB2   = (half_t*)  alloc((size_t)HID1 * OUT_DIM * 2);
    float*    asrc1 = (float*)   alloc((size_t)N * HEADS * 4);
    float*    adst1 = (float*)   alloc((size_t)N * HEADS * 4);
    unsigned* m1    = (unsigned*)alloc((size_t)N * HEADS * 4);
    float*    s1    = (float*)   alloc((size_t)N * HEADS * 4);
    float*    asrc2 = (float*)   alloc((size_t)N * 4);
    float*    adst2 = (float*)   alloc((size_t)N * 4);
    unsigned* m2    = (unsigned*)alloc((size_t)N * 4);
    float*    s2    = (float*)   alloc((size_t)N * 4);
    half_t*   pA2   = (half_t*)h1;   // alias: h1 dead after edge_agg layer 1

    auto cdiv = [](long long a, long long b) { return (int)((a + b - 1) / b); };
    const int B = 256;

    // ---- init accumulators ----
    hipMemsetAsync(s1,   0, (size_t)N * HEADS * 4, stream);
    hipMemsetAsync(agg1, 0, (size_t)N * HID1 * 4, stream);
    hipMemsetAsync(s2,   0, (size_t)N * 4, stream);
    hipMemsetAsync(out,  0, (size_t)N * OUT_DIM * 4, stream);
    fill_u32<<<cdiv((long long)N * HEADS, B), B, 0, stream>>>(m1, NEG_INF_KEY, N * HEADS);
    fill_u32<<<cdiv(N, B), B, 0, stream>>>(m2, NEG_INF_KEY, N);

    // ---- pack operands into WMMA fragment order ----
    pack_a_f32<<<cdiv((long long)N * IN_DIM, B), B, 0, stream>>>(x, pA1, N, IN_DIM);
    pack_b_f32<<<cdiv(IN_DIM * HID1, B), B, 0, stream>>>(W1, pB1, IN_DIM, HID1);
    pack_b_f32<<<cdiv(HID1 * OUT_DIM, B), B, 0, stream>>>(W2, pB2, HID1, OUT_DIM);

    // ---- layer 1: WMMA GEMM (K=256 -> NKT=8; 16 col tiles -> 4 waves x CT=4) ----
    wmma_gemm_packed<8, 4><<<N / 16, 128, 0, stream>>>(pA1, pB1, h1, HID1);
    node_alpha<<<cdiv((long long)N * HEADS, B), B, 0, stream>>>(h1, asv1, adv1, asrc1, adst1, N, HEADS, HID);
    edge_max   <<<cdiv((long long)ET * HEADS, B), B, 0, stream>>>(ei, asrc1, adst1, m1, E, N, HEADS);
    edge_expsum<<<cdiv((long long)ET * HEADS, B), B, 0, stream>>>(ei, asrc1, adst1, m1, s1, E, N, HEADS);
    edge_agg<8, 32><<<cdiv((long long)ET * 32, B), B, 0, stream>>>(ei, asrc1, adst1, m1, s1, h1, agg1, E, N);
    // epilogue writes GEMM2's A operand directly in packed fragment order
    elu_bias_pack<<<cdiv((long long)N * HID1, B), B, 0, stream>>>(agg1, b1, pA2, N, HID1);

    // ---- layer 2: WMMA GEMM (K=256 -> NKT=8; 4 col tiles -> 1 wave x CT=4) ----
    wmma_gemm_packed<8, 4><<<N / 16, 32, 0, stream>>>(pA2, pB2, g2, OUT_DIM);
    node_alpha<<<cdiv(N, B), B, 0, stream>>>(g2, asv2, adv2, asrc2, adst2, N, 1, OUT_DIM);
    edge_max   <<<cdiv(ET, B), B, 0, stream>>>(ei, asrc2, adst2, m2, E, N, 1);
    edge_expsum<<<cdiv(ET, B), B, 0, stream>>>(ei, asrc2, adst2, m2, s2, E, N, 1);
    edge_agg<1, 64><<<cdiv((long long)ET * 32, B), B, 0, stream>>>(ei, asrc2, adst2, m2, s2, g2, out, E, N);
    bias_add<<<cdiv((long long)N * OUT_DIM, B), B, 0, stream>>>(out, b2, N * OUT_DIM, OUT_DIM);
}
